// SSMXLSTMFusion_78709570667253
// MI455X (gfx1250) — matrix operand order repaired and checked by reading, not compiled, not run on hardware
//
#include <hip/hip_runtime.h>
#include <cstdint>

#define DD   512
#define BB   256
#define SS   3
#define TAYLOR_K 36

typedef __attribute__((ext_vector_type(16))) _Float16 v16h;
typedef __attribute__((ext_vector_type(8)))  float    v8f;

union HFrag { uint4 q[2]; v16h v; };

// ---------------------------------------------------------------- conversions
__global__ void cvt_f32_f16(const float* __restrict__ src, _Float16* __restrict__ dst, int n) {
  int i = blockIdx.x * blockDim.x + threadIdx.x;
  if (i < n) dst[i] = (_Float16)src[i];
}

__global__ void cvt2d_f32_f16(const float* __restrict__ src, int lds,
                              _Float16* __restrict__ dst, int ldd, int rows, int cols) {
  int i = blockIdx.x * blockDim.x + threadIdx.x;
  if (i >= rows * cols) return;
  int r = i / cols, c = i - r * cols;
  dst[(size_t)r * ldd + c] = (_Float16)src[(size_t)r * lds + c];
}

// ---------------------------------------------------------------- WMMA GEMM
// C[M,N] (f32, ldC) (+)= X[M,K] (f16, ldX) * W[N,K]^T (f16, ldW) + bias[N]
// Wave tile: 32x32 (2x2 of 16x16 WMMA tiles) -> 4 WMMAs per 8 b128 loads per K-step.
// Block: 128 threads = 4 waves along N (block tile 32M x 128N).
// Requires: M % 32 == 0, N % 128 == 0, K % 32 == 0.
__global__ void wmma_gemm_kernel(const _Float16* __restrict__ X, int ldX,
                                 const _Float16* __restrict__ W, int ldW,
                                 float* __restrict__ C, int ldC,
                                 const float* __restrict__ bias,
                                 int M, int N, int K, int accumulate) {
  const int lane = threadIdx.x & 31;
  const int wave = threadIdx.x >> 5;
  const int tn0 = (blockIdx.x * 4 + wave) * 32;
  const int tm0 = blockIdx.y * 32;
  if (tn0 >= N || tm0 >= M) return;

  const int r  = lane & 15;
  const int hi = lane >> 4;   // 0 or 1

  // A fragment (16x32 f16): lane<16 -> K = {0..7,16..23}; lane>=16 -> K = {8..15,24..31}
  const _Float16* xrow0 = X + (size_t)(tm0 + r)      * ldX + hi * 8;
  const _Float16* xrow1 = X + (size_t)(tm0 + 16 + r) * ldX + hi * 8;
  // B fragment (32x16 f16): lane<16 -> K = {0..7,16..23}; lane>=16 -> K = {8..15,24..31}
  const _Float16* wrow0 = W + (size_t)(tn0 + r)      * ldW + hi * 8;
  const _Float16* wrow1 = W + (size_t)(tn0 + 16 + r) * ldW + hi * 8;

  v8f acc00 = {}, acc01 = {}, acc10 = {}, acc11 = {};
  for (int kk = 0; kk < K; kk += 32) {
    HFrag a0, a1, b0, b1;
    a0.q[0] = *(const uint4*)(xrow0 + kk);
    a0.q[1] = *(const uint4*)(xrow0 + kk + 16);
    a1.q[0] = *(const uint4*)(xrow1 + kk);
    a1.q[1] = *(const uint4*)(xrow1 + kk + 16);
    b0.q[0] = *(const uint4*)(wrow0 + kk);
    b0.q[1] = *(const uint4*)(wrow0 + kk + 16);
    b1.q[0] = *(const uint4*)(wrow1 + kk);
    b1.q[1] = *(const uint4*)(wrow1 + kk + 16);
    if (kk + 32 < K) {
      __builtin_prefetch(xrow0 + kk + 32, 0, 3);   // global_prefetch_b8, WGP scope
      __builtin_prefetch(xrow1 + kk + 32, 0, 3);
      __builtin_prefetch(wrow0 + kk + 32, 0, 3);
      __builtin_prefetch(wrow1 + kk + 32, 0, 3);
    }
    acc00 = __builtin_amdgcn_wmma_f32_16x16x32_f16(false, a0.v, false, b0.v,
                                                   (short)0, acc00, false, false);
    acc01 = __builtin_amdgcn_wmma_f32_16x16x32_f16(false, a0.v, false, b1.v,
                                                   (short)0, acc01, false, false);
    acc10 = __builtin_amdgcn_wmma_f32_16x16x32_f16(false, a1.v, false, b0.v,
                                                   (short)0, acc10, false, false);
    acc11 = __builtin_amdgcn_wmma_f32_16x16x32_f16(false, a1.v, false, b1.v,
                                                   (short)0, acc11, false, false);
  }

  // D layout: VGPR r8, lanes0-15 -> M=r8, N=lane; lanes16-31 -> M=r8+8, N=lane-16
  const int col0 = tn0 + r;
  const int col1 = tn0 + 16 + r;
  const float bv0 = bias ? bias[col0] : 0.0f;
  const float bv1 = bias ? bias[col1] : 0.0f;
  const int rowb0 = tm0 + 8 * hi;
  const int rowb1 = tm0 + 16 + 8 * hi;
#pragma unroll
  for (int r8 = 0; r8 < 8; ++r8) {
    size_t i00 = (size_t)(rowb0 + r8) * ldC + col0;
    size_t i01 = (size_t)(rowb0 + r8) * ldC + col1;
    size_t i10 = (size_t)(rowb1 + r8) * ldC + col0;
    size_t i11 = (size_t)(rowb1 + r8) * ldC + col1;
    float v00 = acc00[r8] + bv0;
    float v01 = acc01[r8] + bv1;
    float v10 = acc10[r8] + bv0;
    float v11 = acc11[r8] + bv1;
    if (accumulate) { v00 += C[i00]; v01 += C[i01]; v10 += C[i10]; v11 += C[i11]; }
    C[i00] = v00; C[i01] = v01; C[i10] = v10; C[i11] = v11;
  }
}

// ---------------------------------------------------------------- layernorm (+gelu)
__global__ void ln_kernel(const float* __restrict__ in, const float* __restrict__ g,
                          const float* __restrict__ beta, float* __restrict__ out,
                          int cols, int do_gelu) {
  __shared__ float s1[256], s2[256];
  const int row = blockIdx.x, tid = threadIdx.x;
  const float* ip = in + (size_t)row * cols;
  float s = 0.f, ss = 0.f;
  for (int c = tid; c < cols; c += 256) { float v = ip[c]; s += v; ss += v * v; }
  s1[tid] = s; s2[tid] = ss; __syncthreads();
  for (int off = 128; off > 0; off >>= 1) {
    if (tid < off) { s1[tid] += s1[tid + off]; s2[tid] += s2[tid + off]; }
    __syncthreads();
  }
  const float mean = s1[0] / cols;
  const float var  = s2[0] / cols - mean * mean;
  const float rstd = rsqrtf(var + 1e-5f);
  float* op = out + (size_t)row * cols;
  for (int c = tid; c < cols; c += 256) {
    float v = (ip[c] - mean) * rstd * g[c] + beta[c];
    if (do_gelu) v = 0.5f * v * (1.0f + erff(v * 0.70710678f));
    op[c] = v;
  }
}

// ---------------------------------------------------------------- delta = softplus(h1 . w2 + b2)
__global__ void delta_kernel(const float* __restrict__ h1, const float* __restrict__ w2,
                             const float* __restrict__ b2, float* __restrict__ delta) {
  __shared__ float s1[256];
  const int row = blockIdx.x, tid = threadIdx.x;
  const float* ip = h1 + (size_t)row * DD;
  float s = 0.f;
  for (int c = tid; c < DD; c += 256) s += ip[c] * w2[c];
  s1[tid] = s; __syncthreads();
  for (int off = 128; off > 0; off >>= 1) {
    if (tid < off) s1[tid] += s1[tid + off];
    __syncthreads();
  }
  if (tid == 0) {
    float xv = s1[0] + b2[0];
    delta[row] = (xv > 20.0f) ? xv : log1pf(expf(xv));
  }
}

// ---------------------------------------------------------------- SSM Taylor helpers
__global__ void ssm_init(const float* __restrict__ h_prev, float* __restrict__ acc,
                         _Float16* __restrict__ Vh) {
  int i = blockIdx.x * blockDim.x + threadIdx.x;
  if (i >= BB * DD) return;
  acc[i] = h_prev[i];
  Vh[i]  = (_Float16)h_prev[i];
}

// Vnew = V_k / 2^(k-1); acc += (delta^k * 2^(k-1) / k!) * Vnew; Vh = Vnew/2 (= V_k/2^k)
__global__ void ssm_axpy(const float* __restrict__ Vnew, const float* __restrict__ delta,
                         float* __restrict__ acc, _Float16* __restrict__ Vh, int k) {
  int i = blockIdx.x * blockDim.x + threadIdx.x;
  if (i >= BB * DD) return;
  int b = i >> 9;  // /512
  float d = delta[b];
  float coef = expf((float)k * logf(d) + (float)(k - 1) * 0.69314718f - lgammaf((float)k + 1.0f));
  acc[i] += coef * Vnew[i];
  Vh[i] = (_Float16)(0.5f * Vnew[i]);
}

// h_ssm = acc + delta * Bx ; also f16 copy into combined slice 0
__global__ void hssm_final(const float* __restrict__ acc, const float* __restrict__ Bx,
                           const float* __restrict__ delta, float* __restrict__ hssm_out,
                           _Float16* __restrict__ combined) {
  int i = blockIdx.x * blockDim.x + threadIdx.x;
  if (i >= BB * DD) return;
  int b = i >> 9, c = i & 511;
  float v = acc[i] + delta[b] * Bx[i];
  hssm_out[i] = v;
  combined[(size_t)b * (5 * DD) + c] = (_Float16)v;
}

// ---------------------------------------------------------------- LSTM pointwise
__global__ void lstm_pw(const float* __restrict__ gates,    // (B, 3*2048), ld = 6144
                        const float* __restrict__ c_in,     // (S,B,D)
                        const float* __restrict__ bih, const float* __restrict__ bhh, // (S,2048)
                        const float* __restrict__ decays,
                        float* __restrict__ h_out, float* __restrict__ c_out, // (S,B,D)
                        _Float16* __restrict__ hnb,          // (S,B,D) f16
                        _Float16* __restrict__ combined) {   // (B, 5D) f16
  int idx = blockIdx.x * blockDim.x + threadIdx.x;
  if (idx >= SS * BB * DD) return;
  int s = idx / (BB * DD);
  int rem = idx - s * (BB * DD);
  int b = rem >> 9, d = rem & 511;
  const float* g  = gates + (size_t)b * (SS * 4 * DD) + s * (4 * DD);
  const float* bi = bih + s * (4 * DD);
  const float* bh = bhh + s * (4 * DD);
  float gi = g[d]          + bi[d]          + bh[d];
  float gf = g[DD + d]     + bi[DD + d]     + bh[DD + d];
  float gg = g[2 * DD + d] + bi[2 * DD + d] + bh[2 * DD + d];
  float go = g[3 * DD + d] + bi[3 * DD + d] + bh[3 * DD + d];
  float ig = 1.0f / (1.0f + expf(-gi));
  float fg = 1.0f / (1.0f + expf(-gf));
  float gv = tanhf(gg);
  float og = 1.0f / (1.0f + expf(-go));
  float c_old = c_in[idx];
  float c_raw = fg * c_old + ig * gv;
  float h = og * tanhf(c_raw);
  float dec = decays[s];
  h_out[idx] = h;
  c_out[idx] = dec * c_old + (1.0f - dec) * c_raw;
  hnb[idx] = (_Float16)h;
  combined[(size_t)b * (5 * DD) + (2 + s) * DD + d] = (_Float16)h;
}

// ---------------------------------------------------------------- attention (seq len S=3)
__global__ void attn_kernel(const float* __restrict__ q,   // (B, D)
                            const float* __restrict__ kb,  // (S*B, D)
                            const float* __restrict__ vb,  // (S*B, D)
                            _Float16* __restrict__ ctxh) { // (B, D) f16
  const int lane = threadIdx.x & 31;
  const int wid  = blockIdx.x * (blockDim.x >> 5) + (threadIdx.x >> 5);
  if (wid >= BB * 8) return;
  const int b = wid >> 3, h = wid & 7;
  const float* qp = q + (size_t)b * DD + h * 64;
  const float q0 = qp[2 * lane], q1 = qp[2 * lane + 1];
  float sc[SS];
#pragma unroll
  for (int s = 0; s < SS; ++s) {
    const float* kp = kb + (size_t)(s * BB + b) * DD + h * 64;
    float p = q0 * kp[2 * lane] + q1 * kp[2 * lane + 1];
#pragma unroll
    for (int m = 16; m >= 1; m >>= 1) p += __shfl_xor(p, m, 32);
    sc[s] = p * 0.125f;   // 1/sqrt(64)
  }
  float mx = fmaxf(sc[0], fmaxf(sc[1], sc[2]));
  float e0 = expf(sc[0] - mx), e1 = expf(sc[1] - mx), e2 = expf(sc[2] - mx);
  float inv = 1.0f / (e0 + e1 + e2);
  float a[SS] = { e0 * inv, e1 * inv, e2 * inv };
  float c0 = 0.f, c1 = 0.f;
#pragma unroll
  for (int s = 0; s < SS; ++s) {
    const float* vp = vb + (size_t)(s * BB + b) * DD + h * 64;
    c0 += a[s] * vp[2 * lane];
    c1 += a[s] * vp[2 * lane + 1];
  }
  ctxh[(size_t)b * DD + h * 64 + 2 * lane]     = (_Float16)c0;
  ctxh[(size_t)b * DD + h * 64 + 2 * lane + 1] = (_Float16)c1;
}

// ---------------------------------------------------------------- host
extern "C" void kernel_launch(void* const* d_in, const int* in_sizes, int n_in,
                              void* d_out, int out_size, void* d_ws, size_t ws_size,
                              hipStream_t stream) {
  (void)in_sizes; (void)n_in; (void)out_size; (void)ws_size;

  const float* x        = (const float*)d_in[0];
  const float* h_prev   = (const float*)d_in[1];
  const float* lstm_h   = (const float*)d_in[2];
  const float* lstm_c   = (const float*)d_in[3];
  const float* A        = (const float*)d_in[4];
  const float* Bm       = (const float*)d_in[5];
  const float* dn_w1    = (const float*)d_in[6];
  const float* dn_b1    = (const float*)d_in[7];
  const float* dn_g     = (const float*)d_in[8];
  const float* dn_beta  = (const float*)d_in[9];
  const float* dn_w2    = (const float*)d_in[10];
  const float* dn_b2    = (const float*)d_in[11];
  const float* wih      = (const float*)d_in[12];
  const float* whh      = (const float*)d_in[13];
  const float* bih      = (const float*)d_in[14];
  const float* bhh      = (const float*)d_in[15];
  const float* decays   = (const float*)d_in[16];
  const float* attn_in_w  = (const float*)d_in[17];
  const float* attn_in_b  = (const float*)d_in[18];
  const float* attn_out_w = (const float*)d_in[19];
  const float* attn_out_b = (const float*)d_in[20];
  const float* proj_w   = (const float*)d_in[21];
  const float* proj_b   = (const float*)d_in[22];
  const float* proj_g   = (const float*)d_in[23];
  const float* proj_beta= (const float*)d_in[24];

  float* out_ln   = (float*)d_out;             // (256,512)
  float* out_hssm = out_ln + BB * DD;          // (256,512)
  float* out_hnew = out_hssm + BB * DD;        // (3,256,512)
  float* out_cnew = out_hnew + SS * BB * DD;   // (3,256,512)

  // ---- workspace arena
  char* p = (char*)d_ws;
  auto alloc = [&](size_t bytes) { void* r = (void*)p; p += (bytes + 255) & ~(size_t)255; return r; };

  _Float16* xh      = (_Float16*)alloc((size_t)BB * DD * 2);
  _Float16* dnw1h   = (_Float16*)alloc((size_t)DD * DD * 2);
  _Float16* Ah      = (_Float16*)alloc((size_t)DD * DD * 2);
  _Float16* Bmh     = (_Float16*)alloc((size_t)DD * DD * 2);
  _Float16* wihh    = (_Float16*)alloc((size_t)SS * 4 * DD * DD * 2);
  _Float16* whhh    = (_Float16*)alloc((size_t)SS * 4 * DD * DD * 2);
  _Float16* attninh = (_Float16*)alloc((size_t)3 * DD * DD * 2);
  _Float16* attnouth= (_Float16*)alloc((size_t)DD * DD * 2);
  _Float16* projh   = (_Float16*)alloc((size_t)DD * 5 * DD * 2);
  _Float16* lstmhh  = (_Float16*)alloc((size_t)SS * BB * DD * 2);
  _Float16* Vh      = (_Float16*)alloc((size_t)BB * DD * 2);
  _Float16* hnb     = (_Float16*)alloc((size_t)SS * BB * DD * 2);
  _Float16* combined= (_Float16*)alloc((size_t)BB * 5 * DD * 2);
  _Float16* ctxh    = (_Float16*)alloc((size_t)BB * DD * 2);

  float* dnout  = (float*)alloc((size_t)BB * DD * 4);
  float* h1     = (float*)alloc((size_t)BB * DD * 4);
  float* delta  = (float*)alloc((size_t)BB * 4);
  float* Vnew   = (float*)alloc((size_t)BB * DD * 4);
  float* accb   = (float*)alloc((size_t)BB * DD * 4);
  float* Bx     = (float*)alloc((size_t)BB * DD * 4);
  float* gates  = (float*)alloc((size_t)BB * SS * 4 * DD * 4);
  float* qbuf   = (float*)alloc((size_t)BB * DD * 4);
  float* kbuf   = (float*)alloc((size_t)SS * BB * DD * 4);
  float* vbuf   = (float*)alloc((size_t)SS * BB * DD * 4);
  float* fuse   = (float*)alloc((size_t)BB * DD * 4);
  float* projout= (float*)alloc((size_t)BB * DD * 4);

  auto cvt = [&](const float* s, _Float16* d, int n) {
    cvt_f32_f16<<<(n + 255) / 256, 256, 0, stream>>>(s, d, n);
  };
  auto gemm = [&](const _Float16* X, int ldX, const _Float16* W, int ldW,
                  float* C, int ldC, const float* bias, int M, int N, int K, int accum) {
    dim3 g(N / 128, M / 32);
    wmma_gemm_kernel<<<g, 128, 0, stream>>>(X, ldX, W, ldW, C, ldC, bias, M, N, K, accum);
  };

  // ---- weight / input conversion to f16
  cvt(x,          xh,      BB * DD);
  cvt(dn_w1,      dnw1h,   DD * DD);
  cvt(A,          Ah,      DD * DD);
  cvt(Bm,         Bmh,     DD * DD);
  cvt(wih,        wihh,    SS * 4 * DD * DD);
  cvt(whh,        whhh,    SS * 4 * DD * DD);
  cvt(attn_in_w,  attninh, 3 * DD * DD);
  cvt(attn_out_w, attnouth, DD * DD);
  cvt(proj_w,     projh,   DD * 5 * DD);
  cvt(lstm_h,     lstmhh,  SS * BB * DD);

  // ---- delta network: h1 = gelu(LN(x @ dn_w1^T + b1)); delta = softplus(h1 @ w2^T + b2)
  gemm(xh, DD, dnw1h, DD, dnout, DD, dn_b1, BB, DD, DD, 0);
  ln_kernel<<<BB, 256, 0, stream>>>(dnout, dn_g, dn_beta, h1, DD, 1);
  delta_kernel<<<BB, 256, 0, stream>>>(h1, dn_w2, dn_b2, delta);

  // ---- SSM: h_ssm = expm(delta*A) h_prev + delta * (Bm x), via scaled Taylor series
  ssm_init<<<(BB * DD + 255) / 256, 256, 0, stream>>>(h_prev, accb, Vh);
  for (int k = 1; k <= TAYLOR_K; ++k) {
    gemm(Vh, DD, Ah, DD, Vnew, DD, nullptr, BB, DD, DD, 0);   // V' = V @ A^T
    ssm_axpy<<<(BB * DD + 255) / 256, 256, 0, stream>>>(Vnew, delta, accb, Vh, k);
  }
  gemm(xh, DD, Bmh, DD, Bx, DD, nullptr, BB, DD, DD, 0);
  hssm_final<<<(BB * DD + 255) / 256, 256, 0, stream>>>(accb, Bx, delta, out_hssm, combined);

  // ---- LSTM gates: one wide GEMM over all 3 layers for x, then per-layer h accumulate
  gemm(xh, DD, wihh, DD, gates, SS * 4 * DD, nullptr, BB, SS * 4 * DD, DD, 0);
  for (int s = 0; s < SS; ++s) {
    gemm(lstmhh + (size_t)s * BB * DD, DD,
         whhh + (size_t)s * 4 * DD * DD, DD,
         gates + (size_t)s * 4 * DD, SS * 4 * DD, nullptr, BB, 4 * DD, DD, 1);
  }
  lstm_pw<<<(SS * BB * DD + 255) / 256, 256, 0, stream>>>(
      gates, lstm_c, bih, bhh, decays, out_hnew, out_cnew, hnb, combined);

  // ---- attention projections (Wq/Wk/Wv rows of attn_in_w)
  gemm(combined, 5 * DD, attninh, DD, qbuf, DD, attn_in_b, BB, DD, DD, 0);  // q from h_ssm slice
  gemm(hnb, DD, attninh + (size_t)DD * DD,     DD, kbuf, DD, attn_in_b + DD,     SS * BB, DD, DD, 0);
  gemm(hnb, DD, attninh + (size_t)2 * DD * DD, DD, vbuf, DD, attn_in_b + 2 * DD, SS * BB, DD, DD, 0);
  attn_kernel<<<BB, 256, 0, stream>>>(qbuf, kbuf, vbuf, ctxh);
  gemm(ctxh, DD, attnouth, DD, fuse, DD, attn_out_b, BB, DD, DD, 0);
  cvt2d_f32_f16<<<(BB * DD + 255) / 256, 256, 0, stream>>>(fuse, DD, combined + DD, 5 * DD, BB, DD);

  // ---- final projection + layernorm
  gemm(combined, 5 * DD, projh, 5 * DD, projout, DD, proj_b, BB, DD, 5 * DD, 0);
  ln_kernel<<<BB, 256, 0, stream>>>(projout, proj_g, proj_beta, out_ln, DD, 0);
}
